// MoELayer_31808527794226
// MI455X (gfx1250) — matrix operand order repaired and checked by reading, not compiled, not run on hardware
//
#include <hip/hip_runtime.h>
#include <math.h>

#define EMB   1024
#define FFN   3072
#define NE    8
#define NTOK  2048
#define NPAIR 4096
#define TILE  32
#define KB    64
#define FC    128
#define DC    128

typedef __attribute__((ext_vector_type(16))) __bf16 v16bf;
typedef __attribute__((ext_vector_type(8)))  float  v8f;

// f32 -> bf16 round-to-nearest-even (bit pattern)
__device__ __forceinline__ unsigned f2bf(float f) {
    unsigned u = __float_as_uint(f);
    u = u + 0x7fffu + ((u >> 16) & 1u);
    return u >> 16;
}
// packed pair via single-instruction hardware convert (RNE)
__device__ __forceinline__ unsigned pack2bf(float lo, float hi) {
    unsigned r;
    asm("v_cvt_pk_bf16_f32 %0, %1, %2" : "=v"(r) : "v"(lo), "v"(hi));
    return r;
}

// A fragment: 16x32 bf16 sub-tile at k-offset kk; lane L: row M=L&15, chunks at kk+half*8, kk+16+half*8
__device__ __forceinline__ v16bf load_frag_a(const unsigned short* lds, int row, int kk, int half, int stride) {
    union { v16bf v; uint4 q[2]; } u;
    const unsigned short* p = lds + row * stride + kk + half * 8;
    u.q[0] = *(const uint4*)(p);
    u.q[1] = *(const uint4*)(p + 16);
    return u.v;
}
// B fragment: 32x16 bf16; lane L holds column N=ncol (L&15), K at k..k+7 and k+16..k+23 (k=kk+half*8)
__device__ __forceinline__ v16bf load_frag_b(const unsigned short* lds, int k, int ncol, int stride) {
    union { v16bf v; unsigned short h[16]; } u;
    const unsigned short* p = lds + k * stride + ncol;
#pragma unroll
    for (int s = 0; s < 8; ++s) {
        u.h[s] = p[s * stride];
        u.h[8 + s] = p[(16 + s) * stride];
    }
    return u.v;
}
__device__ __forceinline__ v8f wmma_bf16(v16bf a, v16bf b, v8f c) {
    return __builtin_amdgcn_wmma_f32_16x16x32_bf16(false, a, false, b, (short)0, c, false, false);
}
// Async global->LDS 16B copy (ASYNCcnt); low 32 bits of generic shared ptr == LDS addr
__device__ __forceinline__ void async_ld_b128(void* lds_ptr, const void* gptr) {
    unsigned lds_addr = (unsigned)(unsigned long long)lds_ptr;
    asm volatile("global_load_async_to_lds_b128 %0, %1, off"
                 :: "v"(lds_addr), "v"(gptr) : "memory");
}
__device__ __forceinline__ void wait_asynccnt0() {
    asm volatile("s_wait_asynccnt 0x0" ::: "memory");
}
__device__ __forceinline__ void wait_asynccnt1() {   // in-order completion: oldest transfer done
    asm volatile("s_wait_asynccnt 0x1" ::: "memory");
}

__global__ __launch_bounds__(256) void moe_zero(float* __restrict__ out, int n, int* __restrict__ counts) {
    for (int i = blockIdx.x * blockDim.x + threadIdx.x; i < n; i += gridDim.x * blockDim.x)
        out[i] = 0.f;
    if (blockIdx.x == 0 && threadIdx.x < NE) counts[threadIdx.x] = 0;
}

// Gating: one wave per token. 8 logits, softmax, top-2, softmax(top-2), per-expert lists.
__global__ __launch_bounds__(256) void moe_gate(
    const float* __restrict__ x, const float* __restrict__ gate_w,
    float* __restrict__ gw, int* __restrict__ counts,
    int* __restrict__ elist, float* __restrict__ wpair)
{
    const int lane = threadIdx.x & 31;
    const int wave = threadIdx.x >> 5;
    const int n = blockIdx.x * 8 + wave;

    float acc[NE];
#pragma unroll
    for (int e = 0; e < NE; ++e) acc[e] = 0.f;
    for (int i = 0; i < EMB / 32; ++i) {
        int d = lane + i * 32;
        float xv = x[(size_t)n * EMB + d];
#pragma unroll
        for (int e = 0; e < NE; ++e) acc[e] += xv * gate_w[e * EMB + d];
    }
#pragma unroll
    for (int e = 0; e < NE; ++e)
        for (int off = 16; off > 0; off >>= 1)
            acc[e] += __shfl_xor(acc[e], off, 32);

    if (lane == 0) {
        float m = acc[0];
#pragma unroll
        for (int e = 1; e < NE; ++e) m = fmaxf(m, acc[e]);
        float w[NE], s = 0.f;
#pragma unroll
        for (int e = 0; e < NE; ++e) { w[e] = __expf(acc[e] - m); s += w[e]; }
        float inv = __builtin_amdgcn_rcpf(s);
#pragma unroll
        for (int e = 0; e < NE; ++e) { w[e] *= inv; gw[n * NE + e] = w[e]; }

        int e1 = 0; float b1 = w[0];
#pragma unroll
        for (int e = 1; e < NE; ++e) if (w[e] > b1) { b1 = w[e]; e1 = e; }
        int e2 = (e1 == 0) ? 1 : 0; float b2 = w[e2];
#pragma unroll
        for (int e = 0; e < NE; ++e) if (e != e1 && w[e] > b2) { b2 = w[e]; e2 = e; }

        float mm = fmaxf(b1, b2);
        float x1 = __expf(b1 - mm), x2 = __expf(b2 - mm);
        float sw = __builtin_amdgcn_rcpf(x1 + x2);

        int p1 = atomicAdd(&counts[e1], 1);
        elist[e1 * NTOK + p1] = 2 * n;
        wpair[2 * n] = x1 * sw;
        int p2 = atomicAdd(&counts[e2], 1);
        elist[e2 * NTOK + p2] = 2 * n + 1;
        wpair[2 * n + 1] = x2 * sw;
    }
}

// Deterministic aux loss: aux = E * sum_e mean_n(gw[n,e]) * (counts[e]/N)
__global__ __launch_bounds__(256) void moe_aux(
    const float* __restrict__ gw, const int* __restrict__ counts, float* __restrict__ out)
{
    __shared__ float red[256];
    const int tid = threadIdx.x;
    float imp[NE];
#pragma unroll
    for (int e = 0; e < NE; ++e) imp[e] = 0.f;
    for (int n = tid; n < NTOK; n += 256)
#pragma unroll
        for (int e = 0; e < NE; ++e) imp[e] += gw[n * NE + e];

    float aux = 0.f;
    for (int e = 0; e < NE; ++e) {
        red[tid] = imp[e];
        __syncthreads();
        for (int s = 128; s > 0; s >>= 1) {
            if (tid < s) red[tid] += red[tid + s];
            __syncthreads();
        }
        if (tid == 0) {
            float importance = red[0] / (float)NTOK;
            float load = (float)counts[e] / (float)NTOK;
            aux += importance * load;
        }
        __syncthreads();
    }
    if (tid == 0) out[(size_t)NTOK * EMB] = (float)NE * aux;
}

// GEMM1: h[pair, f] = silu(x @ Wg^T) * (x @ Wu^T), routed tokens, bf16 WMMA,
// software-pipelined: next stage's global loads overlap current stage's WMMAs.
__global__ __launch_bounds__(256) void moe_gemm1(
    const float* __restrict__ x, const float* __restrict__ Wg, const float* __restrict__ Wu,
    const int* __restrict__ counts, const int* __restrict__ elist,
    unsigned short* __restrict__ hws)
{
    __shared__ __align__(16) unsigned short Xs[TILE * 72];   // [m][k], 144B row = 9*16
    __shared__ __align__(16) unsigned short Wgs[KB * 136];   // [k=d][n=f]
    __shared__ __align__(16) unsigned short Wus[KB * 136];
    __shared__ int spid[TILE];
    __shared__ int stok[TILE];

    const int e = blockIdx.z;
    const int cnt = counts[e];
    const int tbase = blockIdx.y * TILE;
    if (tbase >= cnt) return;                       // block-uniform exit
    const int nrows = min(TILE, cnt - tbase);
    const int fbase = blockIdx.x * FC;
    const int tid = threadIdx.x;

    if (tid < TILE) {
        int p = (tid < nrows) ? elist[e * NTOK + tbase + tid] : -1;
        spid[tid] = p;
        stok[tid] = (p < 0) ? 0 : (p >> 1);
    }
    __syncthreads();                                // stok valid before prefetch

    const int wave = tid >> 5, lane = tid & 31, half = (tid >> 4) & 1, l15 = tid & 15;
    const int mi = wave & 1;
    const int f0 = (wave >> 1) * 2;                 // f-tiles f0, f0+1

    v8f ag0 = {}, ag1 = {}, au0 = {}, au1 = {};
    const float* wgB = Wg + (size_t)e * FFN * EMB + (size_t)fbase * EMB;
    const float* wuB = Wu + (size_t)e * FFN * EMB + (size_t)fbase * EMB;

    // per-thread staging assignments
    const int xr0 = tid >> 4,         xc0 = (tid & 15) * 4;          // X: j=0
    const int xr1 = (tid + 256) >> 4, xc1 = ((tid + 256) & 15) * 4;  // X: j=1
    const size_t xo0 = (size_t)stok[xr0] * EMB + xc0;
    const size_t xo1 = (size_t)stok[xr1] * EMB + xc1;

    float4 px[2];
    float4 pg0[4], pg1[4], pu0[4], pu1[4];

    // prologue prefetch (kb = 0)
    px[0] = *(const float4*)(x + xo0);
    px[1] = *(const float4*)(x + xo1);
#pragma unroll
    for (int j = 0; j < 4; ++j) {
        int q = tid + j * 256;
        int fp = q >> 4, d4 = (q & 15) * 4;
        const float* gp = wgB + (size_t)(2 * fp) * EMB + d4;
        const float* up = wuB + (size_t)(2 * fp) * EMB + d4;
        pg0[j] = *(const float4*)(gp);
        pg1[j] = *(const float4*)(gp + EMB);
        pu0[j] = *(const float4*)(up);
        pu1[j] = *(const float4*)(up + EMB);
    }

    for (int kb = 0; kb < EMB; kb += KB) {
        __syncthreads();                            // previous compute done reading LDS
        {   // store prefetched X tile
            uint2 pk;
            pk.x = pack2bf(px[0].x, px[0].y); pk.y = pack2bf(px[0].z, px[0].w);
            *(uint2*)(Xs + xr0 * 72 + xc0) = pk;
            pk.x = pack2bf(px[1].x, px[1].y); pk.y = pack2bf(px[1].z, px[1].w);
            *(uint2*)(Xs + xr1 * 72 + xc1) = pk;
        }
#pragma unroll
        for (int j = 0; j < 4; ++j) {               // store prefetched Wg/Wu, transposed
            int q = tid + j * 256;
            int fp = q >> 4, d4 = (q & 15) * 4;
            *(unsigned*)(Wgs + (d4 + 0) * 136 + 2 * fp) = pack2bf(pg0[j].x, pg1[j].x);
            *(unsigned*)(Wgs + (d4 + 1) * 136 + 2 * fp) = pack2bf(pg0[j].y, pg1[j].y);
            *(unsigned*)(Wgs + (d4 + 2) * 136 + 2 * fp) = pack2bf(pg0[j].z, pg1[j].z);
            *(unsigned*)(Wgs + (d4 + 3) * 136 + 2 * fp) = pack2bf(pg0[j].w, pg1[j].w);
            *(unsigned*)(Wus + (d4 + 0) * 136 + 2 * fp) = pack2bf(pu0[j].x, pu1[j].x);
            *(unsigned*)(Wus + (d4 + 1) * 136 + 2 * fp) = pack2bf(pu0[j].y, pu1[j].y);
            *(unsigned*)(Wus + (d4 + 2) * 136 + 2 * fp) = pack2bf(pu0[j].z, pu1[j].z);
            *(unsigned*)(Wus + (d4 + 3) * 136 + 2 * fp) = pack2bf(pu0[j].w, pu1[j].w);
        }
        __syncthreads();                            // publish LDS tiles

        const int kn = kb + KB;
        if (kn < EMB) {                             // issue next stage's global loads now;
            px[0] = *(const float4*)(x + xo0 + kn); // latency overlaps WMMAs below
            px[1] = *(const float4*)(x + xo1 + kn);
#pragma unroll
            for (int j = 0; j < 4; ++j) {
                int q = tid + j * 256;
                int fp = q >> 4, d4 = (q & 15) * 4;
                const float* gp = wgB + (size_t)(2 * fp) * EMB + kn + d4;
                const float* up = wuB + (size_t)(2 * fp) * EMB + kn + d4;
                pg0[j] = *(const float4*)(gp);
                pg1[j] = *(const float4*)(gp + EMB);
                pu0[j] = *(const float4*)(up);
                pu1[j] = *(const float4*)(up + EMB);
            }
        }
#pragma unroll
        for (int kk = 0; kk < KB; kk += 32) {
            v16bf a   = load_frag_a(Xs, mi * 16 + l15, kk, half, 72);
            v16bf bg0 = load_frag_b(Wgs, kk + half * 8, (f0 + 0) * 16 + l15, 136);
            v16bf bg1 = load_frag_b(Wgs, kk + half * 8, (f0 + 1) * 16 + l15, 136);
            v16bf bu0 = load_frag_b(Wus, kk + half * 8, (f0 + 0) * 16 + l15, 136);
            v16bf bu1 = load_frag_b(Wus, kk + half * 8, (f0 + 1) * 16 + l15, 136);
            ag0 = wmma_bf16(a, bg0, ag0);
            ag1 = wmma_bf16(a, bg1, ag1);
            au0 = wmma_bf16(a, bu0, au0);
            au1 = wmma_bf16(a, bu1, au1);
        }
    }
    // epilogue: h = silu(g)*u -> hws
#pragma unroll
    for (int r = 0; r < 8; ++r) {
        int row = mi * 16 + r + 8 * half;
        if (row < nrows) {
            size_t base = (size_t)spid[row] * FFN + fbase;
            float g0 = ag0[r], u0 = au0[r];
            float h0 = g0 * __builtin_amdgcn_rcpf(1.f + __expf(-g0)) * u0;
            hws[base + (size_t)(f0 + 0) * 16 + l15] = (unsigned short)f2bf(h0);
            float g1 = ag1[r], u1 = au1[r];
            float h1 = g1 * __builtin_amdgcn_rcpf(1.f + __expf(-g1)) * u1;
            hws[base + (size_t)(f0 + 1) * 16 + l15] = (unsigned short)f2bf(h1);
        }
    }
}

// GEMM2: out[token] += weight * (h @ Wd^T); H tile double-buffered via async DMA,
// Wd register-prefetched -- both overlap the WMMA block.
__global__ __launch_bounds__(256) void moe_gemm2(
    const unsigned short* __restrict__ hws, const float* __restrict__ Wd,
    const int* __restrict__ counts, const int* __restrict__ elist,
    const float* __restrict__ wpair, float* __restrict__ out)
{
    __shared__ __align__(16) unsigned short Hs[2][TILE * 72]; // double-buffered [m][k]
    __shared__ __align__(16) unsigned short Wds[KB * 136];    // [k=f][n=d]
    __shared__ int spid[TILE];
    __shared__ int stok[TILE];
    __shared__ float swt[TILE];

    const int e = blockIdx.z;
    const int cnt = counts[e];
    const int tbase = blockIdx.y * TILE;
    if (tbase >= cnt) return;
    const int nrows = min(TILE, cnt - tbase);
    const int dbase = blockIdx.x * DC;
    const int tid = threadIdx.x;

    if (tid < TILE) {
        int p = (tid < nrows) ? elist[e * NTOK + tbase + tid] : -1;
        spid[tid] = (p < 0) ? 0 : p;
        stok[tid] = (p < 0) ? 0 : (p >> 1);
        swt[tid]  = (p < 0) ? 0.f : wpair[p];
    }
    __syncthreads();                                // spid valid before async issue

    const int wave = tid >> 5, lane = tid & 31, half = (tid >> 4) & 1, l15 = tid & 15;
    const int mi = wave & 1;
    const int d0 = (wave >> 1) * 2;
    const int hrow = tid >> 3, hch = tid & 7;       // async copy: 32 rows x 8 x 16B

    v8f a0 = {}, a1 = {};
    const float* wdB = Wd + (size_t)e * EMB * FFN + (size_t)dbase * FFN;
    const unsigned short* hsrc = hws + (size_t)spid[hrow] * FFN + hch * 8;
    unsigned short* hdst = &Hs[0][hrow * 72 + hch * 8];

    float4 pw0[4], pw1[4];

    // prologue: async H tile 0 + Wd prefetch (kb = 0)
    async_ld_b128(hdst, hsrc);
#pragma unroll
    for (int j = 0; j < 4; ++j) {
        int q = tid + j * 256;
        int dp = q >> 4, f4 = (q & 15) * 4;
        const float* wp = wdB + (size_t)(2 * dp) * FFN + f4;
        pw0[j] = *(const float4*)(wp);
        pw1[j] = *(const float4*)(wp + FFN);
    }

    for (int kb = 0; kb < FFN; kb += KB) {
        const int cur = (kb >> 6) & 1;
        __syncthreads();                            // previous compute done reading LDS
#pragma unroll
        for (int j = 0; j < 4; ++j) {               // store prefetched Wd, transposed
            int q = tid + j * 256;
            int dp = q >> 4, f4 = (q & 15) * 4;
            *(unsigned*)(Wds + (f4 + 0) * 136 + 2 * dp) = pack2bf(pw0[j].x, pw1[j].x);
            *(unsigned*)(Wds + (f4 + 1) * 136 + 2 * dp) = pack2bf(pw0[j].y, pw1[j].y);
            *(unsigned*)(Wds + (f4 + 2) * 136 + 2 * dp) = pack2bf(pw0[j].z, pw1[j].z);
            *(unsigned*)(Wds + (f4 + 3) * 136 + 2 * dp) = pack2bf(pw0[j].w, pw1[j].w);
        }
        const int kn = kb + KB;
        if (kn < FFN) {
            // async DMA next H tile into the other buffer; overlaps compute
            async_ld_b128(&Hs[1 - cur][hrow * 72 + hch * 8], hsrc + kn);
#pragma unroll
            for (int j = 0; j < 4; ++j) {           // prefetch next Wd chunk
                int q = tid + j * 256;
                int dp = q >> 4, f4 = (q & 15) * 4;
                const float* wp = wdB + (size_t)(2 * dp) * FFN + kn + f4;
                pw0[j] = *(const float4*)(wp);
                pw1[j] = *(const float4*)(wp + FFN);
            }
            wait_asynccnt1();                       // oldest (current tile) complete
        } else {
            wait_asynccnt0();                       // drain on last stage
        }
        __syncthreads();                            // publish Hs[cur] + Wds
#pragma unroll
        for (int kk = 0; kk < KB; kk += 32) {
            v16bf a  = load_frag_a(Hs[cur], mi * 16 + l15, kk, half, 72);
            v16bf b0 = load_frag_b(Wds, kk + half * 8, (d0 + 0) * 16 + l15, 136);
            v16bf b1 = load_frag_b(Wds, kk + half * 8, (d0 + 1) * 16 + l15, 136);
            a0 = wmma_bf16(a, b0, a0);
            a1 = wmma_bf16(a, b1, a1);
        }
    }
#pragma unroll
    for (int r = 0; r < 8; ++r) {
        int row = mi * 16 + r + 8 * half;
        if (row < nrows) {
            float w = swt[row];
            float* p = out + (size_t)stok[row] * EMB + dbase;
            unsafeAtomicAdd(p + (d0 + 0) * 16 + l15, a0[r] * w);
            unsafeAtomicAdd(p + (d0 + 1) * 16 + l15, a1[r] * w);
        }
    }
}

extern "C" void kernel_launch(void* const* d_in, const int* in_sizes, int n_in,
                              void* d_out, int out_size, void* d_ws, size_t ws_size,
                              hipStream_t stream) {
    (void)in_sizes; (void)n_in; (void)out_size; (void)ws_size;
    const float* x      = (const float*)d_in[0];
    const float* gate_w = (const float*)d_in[1];
    const float* Wg     = (const float*)d_in[2];
    const float* Wu     = (const float*)d_in[3];
    const float* Wd     = (const float*)d_in[4];
    float* out = (float*)d_out;

    char* ws = (char*)d_ws;
    float* gw            = (float*)ws;                     // 2048*8*4   = 65536
    int*   counts        = (int*)(ws + 65536);             // 32 (pad 256)
    int*   elist         = (int*)(ws + 65792);             // 8*2048*4   = 65536
    float* wpair         = (float*)(ws + 131328);          // 4096*4     = 16384
    unsigned short* hws  = (unsigned short*)(ws + 147712); // 4096*3072*2 = 25165824

    moe_zero<<<4096, 256, 0, stream>>>(out, NTOK * EMB, counts);
    moe_gate<<<NTOK / 8, 256, 0, stream>>>(x, gate_w, gw, counts, elist, wpair);
    moe_aux<<<1, 256, 0, stream>>>(gw, counts, out);
    moe_gemm1<<<dim3(FFN / FC, NTOK / TILE, NE), 256, 0, stream>>>(x, Wg, Wu, counts, elist, hws);
    moe_gemm2<<<dim3(EMB / DC, NTOK / TILE, NE), 256, 0, stream>>>(hws, Wd, counts, elist, wpair, out);
}
